// Qwen3OmniMoeTransformerDecoderLayer_56547539419773
// MI455X (gfx1250) — matrix operand order repaired and checked by reading, not compiled, not run on hardware
//
#include <hip/hip_runtime.h>

typedef __bf16 bf16;
typedef __attribute__((ext_vector_type(16))) __bf16 bf16x16;
typedef __attribute__((ext_vector_type(8)))  float   f32x8;
typedef __attribute__((ext_vector_type(4)))  float   f32x4;

union Frag { bf16x16 v; f32x4 q[2]; };

static __device__ __forceinline__ f32x8 zero8() {
  f32x8 z;
#pragma unroll
  for (int i = 0; i < 8; ++i) z[i] = 0.0f;
  return z;
}

static __device__ __forceinline__ f32x8 wmma_bf16(Frag a, Frag b, f32x8 c) {
  return __builtin_amdgcn_wmma_f32_16x16x32_bf16(false, a.v, false, b.v, (short)0, c, false, false);
}

// gfx1250 async global->LDS copy (ASYNCcnt path). lds = wave-relative LDS byte
// offset (low 32 bits of a generic pointer to a __shared__ object).
static __device__ __forceinline__ void async_b128(unsigned lds, const void* g) {
  asm volatile("global_load_async_to_lds_b128 %0, %1, off" :: "v"(lds), "v"(g) : "memory");
}
static __device__ __forceinline__ void wait_async0() {
  asm volatile("s_wait_asynccnt 0x0" ::: "memory");
}
static __device__ __forceinline__ unsigned lds_off(const void* p) {
  return (unsigned)(uintptr_t)p;
}

static __device__ __forceinline__ float waveSum(float v) {
#pragma unroll
  for (int o = 16; o; o >>= 1) v += __shfl_xor(v, o, 32);
  return v;
}
static __device__ __forceinline__ float halfMax16(float v) {
#pragma unroll
  for (int o = 8; o; o >>= 1) v = fmaxf(v, __shfl_xor(v, o, 16));
  return v;
}
static __device__ __forceinline__ float halfSum16(float v) {
#pragma unroll
  for (int o = 8; o; o >>= 1) v += __shfl_xor(v, o, 16);
  return v;
}

// ---------------------------------------------------------------------------
// fp32 -> bf16 elementwise convert (weights, one pass per call)
// ---------------------------------------------------------------------------
__global__ void f32_to_bf16(const float* __restrict__ src, bf16* __restrict__ dst, int n) {
  for (int i = blockIdx.x * 256 + threadIdx.x; i < n; i += gridDim.x * 256)
    dst[i] = (bf16)src[i];
}

// ---------------------------------------------------------------------------
// Row RMSNorm (D=2048) -> bf16
// ---------------------------------------------------------------------------
__global__ __launch_bounds__(256) void rmsnorm_row_bf16(
    const float* __restrict__ x, const float* __restrict__ scale, bf16* __restrict__ out) {
  const int s = blockIdx.x;
  const float* xr = x + (size_t)s * 2048;
  const int tid = threadIdx.x, lane = tid & 31, wave = tid >> 5;
  float v[8];
  float ss = 0.0f;
#pragma unroll
  for (int i = 0; i < 8; ++i) { v[i] = xr[tid + 256 * i]; ss += v[i] * v[i]; }
  ss = waveSum(ss);
  __shared__ float red[8];
  if (lane == 0) red[wave] = ss;
  __syncthreads();
  float tot = 0.0f;
#pragma unroll
  for (int i = 0; i < 8; ++i) tot += red[i];
  const float inv = rsqrtf(tot * (1.0f / 2048.0f) + 1e-6f);
#pragma unroll
  for (int i = 0; i < 8; ++i) {
    int d = tid + 256 * i;
    out[(size_t)s * 2048 + d] = (bf16)(v[i] * inv * scale[d]);
  }
}

// ---------------------------------------------------------------------------
// Generic bf16 WMMA GEMM: C[M,N] = A[M,K] @ B[K,N]
// BM=128 BN=128 BK=32, 256 threads = 8 waves, wave tile 32x64 (2x4 wmma tiles)
// Double-buffered LDS fed by global_load_async_to_lds_b128 (ASYNCcnt pipeline).
// mode 0: Cf = acc
// mode 1: Cf = acc + resid
// mode 2: Cb[r] = bf16(silu(acc + bias)),  A rows gathered via rowidx, r < *cntp
// mode 3: Cf[rowidx[r]] += rowgate[r]*(acc + bias), r < *cntp
// ---------------------------------------------------------------------------
__global__ __launch_bounds__(256) void gemm_bf16_wmma(
    const bf16* __restrict__ A, int lda,
    const bf16* __restrict__ B, int ldb,
    float* __restrict__ Cf, bf16* __restrict__ Cb, int ldc, int Kd,
    int mode,
    const float* __restrict__ resid,
    const float* __restrict__ bias,
    const int* __restrict__ rowidx,
    const float* __restrict__ rowgate,
    const int* __restrict__ cntp) {
  __shared__ bf16 As[2][128][32];
  __shared__ bf16 Bs[2][32][128];
  const int bm = blockIdx.x, bn = blockIdx.y;
  const int cnt = cntp ? *cntp : 0x7fffffff;
  if (bm * 128 >= cnt) return;
  const int tid = threadIdx.x, lane = tid & 31, wave = tid >> 5;
  const int wm = (wave & 3) * 32, wn = (wave >> 2) * 64;

  f32x8 acc[2][4];
#pragma unroll
  for (int i = 0; i < 2; ++i)
#pragma unroll
    for (int j = 0; j < 4; ++j) acc[i][j] = zero8();

  const int arow_l = tid >> 1;
  const int aseg = (tid & 1) * 16;
  const int r_glob = bm * 128 + arow_l;
  int asrc;
  if (mode == 2)      asrc = (r_glob < cnt) ? rowidx[r_glob] : 0;
  else if (mode == 3) asrc = (r_glob < cnt) ? r_glob : 0;
  else                asrc = r_glob;
  const int brow = tid >> 3;
  const int bseg = (tid & 7) * 16;

  const bf16* aBase = A + (size_t)asrc * lda + aseg;
  const bf16* bBase = B + (size_t)brow * ldb + bn * 128 + bseg;
  const int nK = Kd >> 5;

  auto issue = [&](int kt, int b) {
    const bf16* ap = aBase + kt * 32;
    const bf16* bp = bBase + (size_t)(kt * 32) * ldb;
    unsigned la = lds_off(&As[b][arow_l][aseg]);
    unsigned lb = lds_off(&Bs[b][brow][bseg]);
    async_b128(la, ap);
    async_b128(la + 16, ap + 8);
    async_b128(lb, bp);
    async_b128(lb + 16, bp + 8);
  };

  issue(0, 0);
  for (int kt = 0; kt < nK; ++kt) {
    const int b = kt & 1;
    wait_async0();       // this wave's tile-kt async copies done
    __syncthreads();     // everyone's done; also all reads of buf b^1 finished
    if (kt + 1 < nK) {
      issue(kt + 1, b ^ 1);
      __builtin_prefetch(aBase + (kt + 2) * 32, 0, 1);                 // global_prefetch_b8
      __builtin_prefetch(bBase + (size_t)((kt + 2) * 32) * ldb, 0, 1);
    }
    const int mlo = lane & 15, half = lane >> 4;
    Frag af[2], bfr[4];
#pragma unroll
    for (int i = 0; i < 2; ++i) {
      const bf16* p = &As[b][wm + 16 * i + mlo][8 * half];
      af[i].q[0] = *(const f32x4*)p;
      af[i].q[1] = *(const f32x4*)(p + 16);
    }
#pragma unroll
    for (int j = 0; j < 4; ++j) {
#pragma unroll
      for (int s = 0; s < 16; ++s) {
        const int kk = (s < 8) ? 8 * half + s : 8 + 8 * half + s;
        bfr[j].v[s] = Bs[b][kk][wn + 16 * j + mlo];
      }
    }
#pragma unroll
    for (int i = 0; i < 2; ++i)
#pragma unroll
      for (int j = 0; j < 4; ++j)
        acc[i][j] = wmma_bf16(af[i], bfr[j], acc[i][j]);
  }

  const int ncol = lane & 15, rhalf = lane >> 4;
#pragma unroll
  for (int i = 0; i < 2; ++i)
#pragma unroll
    for (int j = 0; j < 4; ++j)
#pragma unroll
      for (int vg = 0; vg < 8; ++vg) {
        const int rl = bm * 128 + wm + 16 * i + vg + 8 * rhalf;
        const int c  = bn * 128 + wn + 16 * j + ncol;
        const float vv = acc[i][j][vg];
        if (mode == 0) {
          Cf[(size_t)rl * ldc + c] = vv;
        } else if (mode == 1) {
          Cf[(size_t)rl * ldc + c] = vv + resid[(size_t)rl * ldc + c];
        } else if (mode == 2) {
          if (rl < cnt) {
            float xx = vv + bias[c];
            Cb[(size_t)rl * ldc + c] = (bf16)(xx / (1.0f + __expf(-xx)));
          }
        } else {
          if (rl < cnt) {
            const int t = rowidx[rl];
            Cf[(size_t)t * ldc + c] += rowgate[rl] * (vv + bias[c]);
          }
        }
      }
}

// ---------------------------------------------------------------------------
// per-head RMSNorm + RoPE, fp32 -> bf16, repack to [H][S][128]
// grid (S, 40): y<32 q heads, 32..35 k heads, 36..39 v heads (convert only)
// ---------------------------------------------------------------------------
__global__ __launch_bounds__(128) void qk_rope_pack(
    const float* __restrict__ qf, const float* __restrict__ kf, const float* __restrict__ vf,
    const float* __restrict__ qscale, const float* __restrict__ kscale,
    bf16* __restrict__ qo, bf16* __restrict__ ko, bf16* __restrict__ vo) {
  const int s = blockIdx.x, y = blockIdx.y, d = threadIdx.x;
  const int lane = d & 31, wave = d >> 5;
  __shared__ float red[4];
  __shared__ float xb[128];
  if (y < 36) {
    float x;
    const float* sc;
    if (y < 32) { x = qf[(size_t)s * 4096 + y * 128 + d]; sc = qscale; }
    else        { x = kf[(size_t)s * 512 + (y - 32) * 128 + d]; sc = kscale; }
    float ss = waveSum(x * x);
    if (lane == 0) red[wave] = ss;
    __syncthreads();
    const float tot = red[0] + red[1] + red[2] + red[3];
    const float xn = x * rsqrtf(tot * (1.0f / 128.0f) + 1e-6f) * sc[d];
    xb[d] = xn;
    __syncthreads();
    const int jj = d & 63;
    const float ang = (float)s * exp2f(-(float)jj * (19.931568569324174f / 64.0f));
    float sn, cs;
    __sincosf(ang, &sn, &cs);
    const float other = (d < 64) ? -xb[d + 64] : xb[d - 64];
    const float r = xn * cs + other * sn;
    if (y < 32) qo[((size_t)y * 2048 + s) * 128 + d] = (bf16)r;
    else        ko[((size_t)(y - 32) * 2048 + s) * 128 + d] = (bf16)r;
  } else {
    vo[((size_t)(y - 36) * 2048 + s) * 128 + d] = (bf16)vf[(size_t)s * 512 + (y - 36) * 128 + d];
  }
}

// ---------------------------------------------------------------------------
// Causal flash attention, bf16 WMMA, GQA 32q/4kv heads, HD=128
// grid (S/128, 32), 256 threads = 8 waves, wave owns 16 q rows
// V tile staged via async global->LDS; K tile transposed through VGPRs.
// ---------------------------------------------------------------------------
__global__ __launch_bounds__(256) void flash_attn(
    const bf16* __restrict__ Q,   // [32][S][128]
    const bf16* __restrict__ Kh,  // [4][S][128]
    const bf16* __restrict__ Vh,  // [4][S][128]
    bf16* __restrict__ O)         // [S][4096]
{
  const int h = blockIdx.y, qb = blockIdx.x;
  const int kh = h >> 3;
  const int lane = threadIdx.x & 31, wave = threadIdx.x >> 5;
  const int q0 = qb * 128 + wave * 16;
  __shared__ bf16 Kt[128][32];       // [dim][key]
  __shared__ bf16 Vt[32][128];       // [key][dim]
  __shared__ bf16 Pst[8][16][32];    // per-wave P staging

  const int mlo = lane & 15, half = lane >> 4;
  Frag aq[4];
  {
    const bf16* qp = Q + ((size_t)h * 2048 + (q0 + mlo)) * 128;
#pragma unroll
    for (int c = 0; c < 4; ++c) {
      aq[c].q[0] = *(const f32x4*)(qp + 32 * c + 8 * half);
      aq[c].q[1] = *(const f32x4*)(qp + 32 * c + 16 + 8 * half);
    }
  }
  f32x8 oacc[8];
#pragma unroll
  for (int t = 0; t < 8; ++t) oacc[t] = zero8();
  float rm[8], rlsum[8];
#pragma unroll
  for (int i = 0; i < 8; ++i) { rm[i] = -1e30f; rlsum[i] = 0.0f; }

  const int jmax = qb * 4 + 3;
  for (int j = 0; j <= jmax; ++j) {
    const int kv = j * 32;
    {  // V tile: async copy, straight layout
      const int key = threadIdx.x >> 3, dseg = (threadIdx.x & 7) * 16;
      const bf16* vp = Vh + ((size_t)kh * 2048 + kv + key) * 128 + dseg;
      unsigned lv = lds_off(&Vt[key][dseg]);
      async_b128(lv, vp);
      async_b128(lv + 16, vp + 8);
    }
    {  // K tile, transposed into LDS
      const int key = threadIdx.x & 31, dseg = threadIdx.x >> 5;
      const bf16* kp = Kh + ((size_t)kh * 2048 + kv + key) * 128 + dseg * 16;
      f32x4 r0 = *(const f32x4*)kp;
      f32x4 r1 = *(const f32x4*)(kp + 8);
      const bf16* e0 = (const bf16*)&r0;
      const bf16* e1 = (const bf16*)&r1;
#pragma unroll
      for (int i = 0; i < 8; ++i) Kt[dseg * 16 + i][key] = e0[i];
#pragma unroll
      for (int i = 0; i < 8; ++i) Kt[dseg * 16 + 8 + i][key] = e1[i];
    }
    wait_async0();
    __syncthreads();

    f32x8 s0 = zero8(), s1 = zero8();
#pragma unroll
    for (int c = 0; c < 4; ++c) {
      Frag b0, b1;
#pragma unroll
      for (int s = 0; s < 16; ++s) {
        const int kk = (s < 8) ? 8 * half + s : 8 + 8 * half + s;
        b0.v[s] = Kt[32 * c + kk][mlo];
        b1.v[s] = Kt[32 * c + kk][16 + mlo];
      }
      s0 = wmma_bf16(aq[c], b0, s0);
      s1 = wmma_bf16(aq[c], b1, s1);
    }
    const float scale = 0.08838834764831845f;  // 1/sqrt(128)
    const int ncol = lane & 15, rhalf = lane >> 4;
    float resc[8];
#pragma unroll
    for (int vg = 0; vg < 8; ++vg) {
      const int qrow = q0 + vg + 8 * rhalf;
      float a = s0[vg] * scale, b = s1[vg] * scale;
      if (kv + ncol > qrow)      a = -1e30f;
      if (kv + 16 + ncol > qrow) b = -1e30f;
      const float mnew = fmaxf(rm[vg], halfMax16(fmaxf(a, b)));
      const float pa = __expf(a - mnew), pb = __expf(b - mnew);
      s0[vg] = pa; s1[vg] = pb;
      const float ls = halfSum16(pa + pb);
      const float rs = __expf(rm[vg] - mnew);
      resc[vg] = rs;
      rlsum[vg] = rlsum[vg] * rs + ls;
      rm[vg] = mnew;
    }
#pragma unroll
    for (int t = 0; t < 8; ++t)
#pragma unroll
      for (int vg = 0; vg < 8; ++vg) oacc[t][vg] *= resc[vg];
#pragma unroll
    for (int vg = 0; vg < 8; ++vg) {   // C-layout -> LDS
      const int mr = vg + 8 * rhalf;
      Pst[wave][mr][ncol]      = (bf16)s0[vg];
      Pst[wave][mr][16 + ncol] = (bf16)s1[vg];
    }
    Frag pA;                            // reload as A-frag (in-order wave LDS)
    pA.q[0] = *(const f32x4*)&Pst[wave][mlo][8 * half];
    pA.q[1] = *(const f32x4*)&Pst[wave][mlo][16 + 8 * half];
#pragma unroll
    for (int t = 0; t < 8; ++t) {
      Frag bv;
#pragma unroll
      for (int s = 0; s < 16; ++s) {
        const int kk = (s < 8) ? 8 * half + s : 8 + 8 * half + s;
        bv.v[s] = Vt[kk][16 * t + mlo];
      }
      oacc[t] = wmma_bf16(pA, bv, oacc[t]);
    }
    __syncthreads();
  }
  const int ncol = lane & 15, rhalf = lane >> 4;
#pragma unroll
  for (int t = 0; t < 8; ++t)
#pragma unroll
    for (int vg = 0; vg < 8; ++vg) {
      const int srow = q0 + vg + 8 * rhalf;
      O[(size_t)srow * 4096 + h * 128 + 16 * t + ncol] = (bf16)(oacc[t][vg] / rlsum[vg]);
    }
}

// ---------------------------------------------------------------------------
// Router: logits, softmax, top-2, normalized gates. aux_loss == 1.0 exactly.
// ---------------------------------------------------------------------------
__global__ __launch_bounds__(256) void router_topk(
    const bf16* __restrict__ xf, const float* __restrict__ rw,
    float* __restrict__ logits, int* __restrict__ sel, float* __restrict__ gate,
    float* __restrict__ aux) {
  const int t = blockIdx.x, tid = threadIdx.x, lane = tid & 31, wave = tid >> 5;
  float acc[8];
#pragma unroll
  for (int e = 0; e < 8; ++e) acc[e] = 0.0f;
  for (int d = tid; d < 2048; d += 256) {
    const float x = (float)xf[(size_t)t * 2048 + d];
    const float* w = rw + (size_t)d * 8;
#pragma unroll
    for (int e = 0; e < 8; ++e) acc[e] += x * w[e];
  }
#pragma unroll
  for (int e = 0; e < 8; ++e) acc[e] = waveSum(acc[e]);
  __shared__ float red[8][8];
  if (lane == 0)
#pragma unroll
    for (int e = 0; e < 8; ++e) red[wave][e] = acc[e];
  __syncthreads();
  if (tid == 0) {
    float lg[8];
    for (int e = 0; e < 8; ++e) {
      float s_ = 0.0f;
      for (int w2 = 0; w2 < 8; ++w2) s_ += red[w2][e];
      lg[e] = s_;
      logits[(size_t)t * 8 + e] = s_;
    }
    float mx = lg[0];
    for (int e = 1; e < 8; ++e) mx = fmaxf(mx, lg[e]);
    float p[8], se = 0.0f;
    for (int e = 0; e < 8; ++e) { p[e] = __expf(lg[e] - mx); se += p[e]; }
    for (int e = 0; e < 8; ++e) p[e] /= se;
    int i0 = 0;
    for (int e = 1; e < 8; ++e) if (p[e] > p[i0]) i0 = e;
    int i1 = (i0 == 0) ? 1 : 0;
    for (int e = 0; e < 8; ++e) if (e != i0 && p[e] > p[i1]) i1 = e;
    const float gs = p[i0] + p[i1];
    sel[t * 2] = i0; sel[t * 2 + 1] = i1;
    gate[t * 2] = p[i0] / gs; gate[t * 2 + 1] = p[i1] / gs;
    if (t == 0) *aux = 1.0f;
  }
}

__global__ void moe_scatter(const int* __restrict__ sel, const float* __restrict__ gate,
                            int* __restrict__ cnt, int* __restrict__ perm,
                            float* __restrict__ pg) {
  const int t = blockIdx.x * 256 + threadIdx.x;
  if (t >= 2048) return;
#pragma unroll
  for (int k2 = 0; k2 < 2; ++k2) {
    const int e = sel[t * 2 + k2];
    const int pos = atomicAdd(&cnt[e], 1);
    perm[e * 2048 + pos] = t;
    pg[e * 2048 + pos] = gate[t * 2 + k2];
  }
}

// ---------------------------------------------------------------------------
extern "C" void kernel_launch(void* const* d_in, const int* in_sizes, int n_in,
                              void* d_out, int out_size, void* d_ws, size_t ws_size,
                              hipStream_t stream) {
  (void)in_sizes; (void)n_in; (void)out_size; (void)ws_size;
  const float* hidden = (const float*)d_in[0];
  const float* ln1s   = (const float*)d_in[1];
  const float* ln2s   = (const float*)d_in[2];
  const float* Wq     = (const float*)d_in[3];
  const float* Wk     = (const float*)d_in[4];
  const float* Wv     = (const float*)d_in[5];
  const float* Wo     = (const float*)d_in[6];
  const float* qns    = (const float*)d_in[7];
  const float* kns    = (const float*)d_in[8];
  const float* rW     = (const float*)d_in[9];
  const float* upW    = (const float*)d_in[10];
  const float* upB    = (const float*)d_in[11];
  const float* dnW    = (const float*)d_in[12];
  const float* dnB    = (const float*)d_in[13];

  float* out_hidden = (float*)d_out;
  float* out_logits = out_hidden + (size_t)2048 * 2048;
  float* out_aux    = out_logits + (size_t)2048 * 8;

  char* w = (char*)d_ws;
  size_t off = 0;
  auto take = [&](size_t bytes) -> void* {
    void* p = w + off;
    off = (off + bytes + 255) & ~(size_t)255;
    return p;
  };
  bf16* wqB  = (bf16*)take((size_t)2048 * 4096 * 2);
  bf16* wkB  = (bf16*)take((size_t)2048 * 512 * 2);
  bf16* wvB  = (bf16*)take((size_t)2048 * 512 * 2);
  bf16* woB  = (bf16*)take((size_t)4096 * 2048 * 2);
  bf16* upWB = (bf16*)take((size_t)8 * 2048 * 768 * 2);
  bf16* dnWB = (bf16*)take((size_t)8 * 768 * 2048 * 2);
  bf16* xn   = (bf16*)take((size_t)2048 * 2048 * 2);
  float* qfb = (float*)take((size_t)2048 * 4096 * 4);
  float* kfb = (float*)take((size_t)2048 * 512 * 4);
  float* vfb = (float*)take((size_t)2048 * 512 * 4);
  bf16* qr   = (bf16*)take((size_t)32 * 2048 * 128 * 2);
  bf16* kr   = (bf16*)take((size_t)4 * 2048 * 128 * 2);
  bf16* vr   = (bf16*)take((size_t)4 * 2048 * 128 * 2);
  bf16* attnB = (bf16*)take((size_t)2048 * 4096 * 2);
  bf16* xfB  = (bf16*)take((size_t)2048 * 2048 * 2);
  int* sel   = (int*)take((size_t)2048 * 2 * 4);
  float* gate = (float*)take((size_t)2048 * 2 * 4);
  int* cnt   = (int*)take(8 * 4);
  int* perm  = (int*)take((size_t)8 * 2048 * 4);
  float* pg  = (float*)take((size_t)8 * 2048 * 4);
  bf16* hbuf = (bf16*)take((size_t)2048 * 768 * 2);

  hipMemsetAsync(cnt, 0, 8 * sizeof(int), stream);

  // Weight conversion fp32 -> bf16
  f32_to_bf16<<<2048, 256, 0, stream>>>(Wq, wqB, 2048 * 4096);
  f32_to_bf16<<<512, 256, 0, stream>>>(Wk, wkB, 2048 * 512);
  f32_to_bf16<<<512, 256, 0, stream>>>(Wv, wvB, 2048 * 512);
  f32_to_bf16<<<2048, 256, 0, stream>>>(Wo, woB, 4096 * 2048);
  f32_to_bf16<<<4096, 256, 0, stream>>>(upW, upWB, 8 * 2048 * 768);
  f32_to_bf16<<<4096, 256, 0, stream>>>(dnW, dnWB, 8 * 768 * 2048);

  // ln1 -> xn (bf16)
  rmsnorm_row_bf16<<<2048, 256, 0, stream>>>(hidden, ln1s, xn);

  // QKV projections
  gemm_bf16_wmma<<<dim3(16, 32), 256, 0, stream>>>(xn, 2048, wqB, 4096, qfb, nullptr, 4096,
      2048, 0, nullptr, nullptr, nullptr, nullptr, nullptr);
  gemm_bf16_wmma<<<dim3(16, 4), 256, 0, stream>>>(xn, 2048, wkB, 512, kfb, nullptr, 512,
      2048, 0, nullptr, nullptr, nullptr, nullptr, nullptr);
  gemm_bf16_wmma<<<dim3(16, 4), 256, 0, stream>>>(xn, 2048, wvB, 512, vfb, nullptr, 512,
      2048, 0, nullptr, nullptr, nullptr, nullptr, nullptr);

  // q/k norm + RoPE + repack
  qk_rope_pack<<<dim3(2048, 40), 128, 0, stream>>>(qfb, kfb, vfb, qns, kns, qr, kr, vr);

  // attention
  flash_attn<<<dim3(16, 32), 256, 0, stream>>>(qr, kr, vr, attnB);

  // Wo projection + residual -> out_hidden
  gemm_bf16_wmma<<<dim3(16, 16), 256, 0, stream>>>(attnB, 4096, woB, 2048, out_hidden,
      nullptr, 2048, 4096, 1, hidden, nullptr, nullptr, nullptr, nullptr);

  // ln2 -> xf (bf16)
  rmsnorm_row_bf16<<<2048, 256, 0, stream>>>(out_hidden, ln2s, xfB);

  // router + scatter
  router_topk<<<2048, 256, 0, stream>>>(xfB, rW, out_logits, sel, gate, out_aux);
  moe_scatter<<<8, 256, 0, stream>>>(sel, gate, cnt, perm, pg);

  // sparse MoE: per-expert gather GEMM (silu) then scatter GEMM (gated accumulate)
  for (int e = 0; e < 8; ++e) {
    gemm_bf16_wmma<<<dim3(16, 6), 256, 0, stream>>>(xfB, 2048, upWB + (size_t)e * 2048 * 768,
        768, nullptr, hbuf, 768, 2048, 2, nullptr, upB + (size_t)e * 768, perm + e * 2048,
        nullptr, cnt + e);
    gemm_bf16_wmma<<<dim3(16, 16), 256, 0, stream>>>(hbuf, 768, dnWB + (size_t)e * 768 * 2048,
        2048, out_hidden, nullptr, 2048, 768, 3, nullptr, dnB + (size_t)e * 2048,
        perm + e * 2048, pg + e * 2048, cnt + e);
  }
}
